// MultiHeadLatentAttention_76630806495588
// MI455X (gfx1250) — matrix operand order repaired and checked by reading, not compiled, not run on hardware
//
#include <hip/hip_runtime.h>
#include <hip/hip_bf16.h>

typedef __attribute__((ext_vector_type(16))) _Float16 v16h;
typedef __attribute__((ext_vector_type(8)))  float    v8f;

union FragU { v16h h; uint4 u[2]; };

__device__ inline v8f wmma16(v16h a, v16h b, v8f c) {
  // v_wmma_f32_16x16x32_f16 : D = A(16x32) * B(32x16) + C(16x16 f32)
  return __builtin_amdgcn_wmma_f32_16x16x32_f16(false, a, false, b, (short)0, c,
                                                false, false);
}

// A fragment 16x32 (MxK) from row-major [M,K]; base points at (row0, k0).
// Lane l: row = l&15, holds K = kb..kb+7 and kb+16..kb+23, kb = (l>>4)*8.
__device__ inline v16h load_frag_a(const _Float16* base, int lda, int lane) {
  const int row = lane & 15;
  const int kb  = (lane >> 4) * 8;
  const _Float16* p = base + (size_t)row * (size_t)lda + kb;
  FragU f;
  f.u[0] = *reinterpret_cast<const uint4*>(p);       // K = kb .. kb+7
  f.u[1] = *reinterpret_cast<const uint4*>(p + 16);  // K = kb+16 .. kb+23
  return f.h;
}

// B fragment 32x16 (KxN) sourced from row-major [N,K]; base at (n0, k0).
// Lane l: col = l&15, holds K = kb..kb+7 and kb+16..kb+23, kb = (l>>4)*8.
__device__ inline v16h load_frag_b(const _Float16* base, int ldb, int lane) {
  const int n  = lane & 15;
  const int kb = (lane >> 4) * 8;
  const _Float16* p = base + (size_t)n * (size_t)ldb + kb;
  FragU f;
  f.u[0] = *reinterpret_cast<const uint4*>(p);
  f.u[1] = *reinterpret_cast<const uint4*>(p + 16);
  return f.h;
}

// ---------------------------------------------------------------------------
// Elementwise f32 -> f16 strided gather: out[((z<<iB)+i)<<jB | j] =
// in[z*sz + i*si + j*sj].  All dims are powers of two -> shift/mask only.
// ---------------------------------------------------------------------------
__global__ void __launch_bounds__(256) cvt_f32_h(
    const float* __restrict__ in, _Float16* __restrict__ out, long long n,
    int iB, int jB, long long sz, long long si, long long sj) {
  long long idx = (long long)blockIdx.x * blockDim.x + threadIdx.x;
  if (idx >= n) return;
  const int j = (int)(idx & ((1LL << jB) - 1));
  const long long t = idx >> jB;
  const int i = (int)(t & ((1LL << iB) - 1));
  const long long z = t >> iB;
  out[idx] = (_Float16)in[z * sz + (long long)i * si + (long long)j * sj];
}

__global__ void __launch_bounds__(256) cvt_h_h(
    const _Float16* __restrict__ in, _Float16* __restrict__ out, long long n,
    int iB, int jB, long long sz, long long si, long long sj) {
  long long idx = (long long)blockIdx.x * blockDim.x + threadIdx.x;
  if (idx >= n) return;
  const int j = (int)(idx & ((1LL << jB) - 1));
  const long long t = idx >> jB;
  const int i = (int)(t & ((1LL << iB) - 1));
  const long long z = t >> iB;
  out[idx] = in[z * sz + (long long)i * si + (long long)j * sj];
}

// ---------------------------------------------------------------------------
// Generic NT GEMM: C[M,N] = sum_g A_g[M,K] * B_g[N,K]^T, both K-fast f16.
// Block = 4 waves; wave computes 16(M) x 64(N); block = 64x64 tile.
// All 10 fragment loads of a K-step issue before the 4 WMMAs (staged waits).
// blockIdx.z -> (b = z/Hdim, h = z%Hdim) with independent strides.
// ---------------------------------------------------------------------------
template <typename OutT>
__global__ void __launch_bounds__(128) gemm_nt(
    const _Float16* __restrict__ A, const _Float16* __restrict__ B,
    OutT* __restrict__ C, int K, int G, long long sAG, long long sBG,
    int lda, int ldb, int ldc, int Hdim,
    long long sAb, long long sAh, long long sBb, long long sBh,
    long long sCb, long long sCh) {
  const int lane = threadIdx.x & 31, wv = threadIdx.x >> 5;
  const int z = blockIdx.z, b = z / Hdim, h = z % Hdim;
  const long long tM = (long long)blockIdx.y * 64 + wv * 16;
  const long long tN = (long long)blockIdx.x * 64;
  const _Float16* Ab = A + sAb * b + sAh * h;
  const _Float16* Bb = B + sBb * b + sBh * h;
  v8f acc[4] = {};
  for (int g = 0; g < G; ++g) {
    const _Float16* Ag = Ab + sAG * g + tM * lda;
    const _Float16* Bg = Bb + sBG * g + tN * ldb;
    for (int k0 = 0; k0 < K; k0 += 32) {
      v16h a  = load_frag_a(Ag + k0, lda, lane);
      v16h b0 = load_frag_b(Bg + k0, ldb, lane);
      v16h b1 = load_frag_b(Bg + 16LL * ldb + k0, ldb, lane);
      v16h b2 = load_frag_b(Bg + 32LL * ldb + k0, ldb, lane);
      v16h b3 = load_frag_b(Bg + 48LL * ldb + k0, ldb, lane);
      __builtin_prefetch(Ag + k0 + 64, 0, 0);  // global_prefetch next K block
      acc[0] = wmma16(a, b0, acc[0]);
      acc[1] = wmma16(a, b1, acc[1]);
      acc[2] = wmma16(a, b2, acc[2]);
      acc[3] = wmma16(a, b3, acc[3]);
    }
  }
  OutT* Cz = C + sCb * b + sCh * h;
  const int grp = lane >> 4, col = lane & 15;
#pragma unroll
  for (int j = 0; j < 4; ++j)
#pragma unroll
    for (int v = 0; v < 8; ++v)
      Cz[(tM + grp * 8 + v) * ldc + tN + j * 16 + col] = (OutT)acc[j][v];
}

// ---------------------------------------------------------------------------
// Flash attention over the latent dim (c = 512), causal, per (b, h, 16 rows).
// Block = 4 waves on the SAME 16 query rows; Q strip staged once in LDS.
//   QK^T: wave w computes S[16 x 16] at t-cols tBase + 16w (K = c = 512),
//         two independent WMMA chains to break the accumulator dependency.
//   softmax: shfl_xor row reductions + LDS-combined running max/sum.
//   P staged in LDS f16 (A-fragment layout); PV: wave w owns c-cols
//   [128w, 128w+128) -> 8 WMMA accumulators vs latT [c,t]; B fragments are
//   loaded in groups of 4 so load latency overlaps the WMMAs.
// ---------------------------------------------------------------------------
__global__ void __launch_bounds__(128) mla_attn(
    const _Float16* __restrict__ qp,    // [b,h,s,c] query_part (c-fast)
    const _Float16* __restrict__ lat,   // [b,t,c]   latents    (c-fast)
    const _Float16* __restrict__ latT,  // [b,c,t]   latents^T  (t-fast)
    _Float16* __restrict__ ctx)         // [b,h,s,c] attn@latents
{
  constexpr int S = 2048, CL = 512;
  const int sT = blockIdx.x, h = blockIdx.y, b = blockIdx.z;
  const int tid = threadIdx.x, lane = tid & 31, wv = tid >> 5;
  const int grp = lane >> 4, col = lane & 15;
  const _Float16* Q  = qp   + ((size_t)(b * 16 + h) * S) * CL;
  const _Float16* Km = lat  + (size_t)b * S * CL;
  const _Float16* Vt = latT + (size_t)b * CL * S;

  __shared__ __align__(16) _Float16 Qs[16][512];  // 16 KB Q strip
  __shared__ __align__(16) _Float16 Pt[16][64];   // 2 KB P tile
  __shared__ float wmax[4][16], wsum[4][16], mrun[16], lrun[16], alf[16];

  const int qrow0 = sT * 16;
  {  // cooperative Q strip load: rows are contiguous -> 1024 x b128
    const uint4* src = reinterpret_cast<const uint4*>(Q + (size_t)qrow0 * CL);
    uint4* dst = reinterpret_cast<uint4*>(&Qs[0][0]);
    for (int i = tid; i < 16 * 512 / 8; i += 128) dst[i] = src[i];
  }
  if (tid < 16) { mrun[tid] = -__builtin_inff(); lrun[tid] = 0.0f; }
  __syncthreads();

  v8f acc[8] = {};  // ctx accumulators: c-cols wv*128 + ct*16 + col
  const int nT = (qrow0 + 15) / 64 + 1;  // causal: t tiles of 64 up to row max
  const float scale = 0.08838834764831845f;  // 1/sqrt(d_head=128)

  for (int tb = 0; tb < nT; ++tb) {
    const int tBase = tb * 64;
    const int tt = tBase + wv * 16;
    v8f s0 = {}, s1 = {};
#pragma unroll 2
    for (int k0 = 0; k0 < CL; k0 += 64) {
      v16h a0 = load_frag_a(&Qs[0][k0], 512, lane);
      v16h b0 = load_frag_b(Km + (size_t)tt * CL + k0, CL, lane);
      v16h a1 = load_frag_a(&Qs[0][k0 + 32], 512, lane);
      v16h b1 = load_frag_b(Km + (size_t)tt * CL + k0 + 32, CL, lane);
      s0 = wmma16(a0, b0, s0);
      s1 = wmma16(a1, b1, s1);
    }
    const v8f sAcc = s0 + s1;
    // scale + causal mask (C layout: row = grp*8+v, col = tt + (lane&15))
    float sv[8];
    const int tg = tt + col;
#pragma unroll
    for (int v = 0; v < 8; ++v) {
      const int mg = qrow0 + grp * 8 + v;
      float xx = sAcc[v] * scale;
      sv[v] = (tg > mg) ? -__builtin_inff() : xx;
    }
    // per-row max across the 16-lane group
#pragma unroll
    for (int v = 0; v < 8; ++v) {
      float m = sv[v];
      m = fmaxf(m, __shfl_xor(m, 1, 32));
      m = fmaxf(m, __shfl_xor(m, 2, 32));
      m = fmaxf(m, __shfl_xor(m, 4, 32));
      m = fmaxf(m, __shfl_xor(m, 8, 32));
      if (col == 0) wmax[wv][grp * 8 + v] = m;
    }
    __syncthreads();
    if (tid < 16) {
      float nm = mrun[tid];
      nm = fmaxf(nm, wmax[0][tid]); nm = fmaxf(nm, wmax[1][tid]);
      nm = fmaxf(nm, wmax[2][tid]); nm = fmaxf(nm, wmax[3][tid]);
      alf[tid]  = __expf(mrun[tid] - nm);  // rescale factor (0 on first tile)
      mrun[tid] = nm;
    }
    __syncthreads();
    // P = exp(S - m_new), row sums, stage P tile in LDS (f16)
#pragma unroll
    for (int v = 0; v < 8; ++v) {
      const int r = grp * 8 + v;
      float p = __expf(sv[v] - mrun[r]);
      float su = p;
      su += __shfl_xor(su, 1, 32);
      su += __shfl_xor(su, 2, 32);
      su += __shfl_xor(su, 4, 32);
      su += __shfl_xor(su, 8, 32);
      if (col == 0) wsum[wv][r] = su;
      Pt[r][wv * 16 + col] = (_Float16)p;
    }
    __syncthreads();
    if (tid < 16)
      lrun[tid] = lrun[tid] * alf[tid] +
                  wsum[0][tid] + wsum[1][tid] + wsum[2][tid] + wsum[3][tid];
    // rescale running ctx, then accumulate P @ latents via WMMA
#pragma unroll
    for (int ct = 0; ct < 8; ++ct)
#pragma unroll
      for (int v = 0; v < 8; ++v) acc[ct][v] *= alf[grp * 8 + v];
#pragma unroll
    for (int t2 = 0; t2 < 64; t2 += 32) {
      v16h pa = load_frag_a(&Pt[0][t2], 64, lane);  // ds_load path
#pragma unroll
      for (int cb = 0; cb < 8; cb += 4) {
        const _Float16* Vb = Vt + (size_t)(wv * 128 + cb * 16) * S + tBase + t2;
        v16h vb0 = load_frag_b(Vb, S, lane);
        v16h vb1 = load_frag_b(Vb + 16LL * S, S, lane);
        v16h vb2 = load_frag_b(Vb + 32LL * S, S, lane);
        v16h vb3 = load_frag_b(Vb + 48LL * S, S, lane);
        acc[cb + 0] = wmma16(pa, vb0, acc[cb + 0]);
        acc[cb + 1] = wmma16(pa, vb1, acc[cb + 1]);
        acc[cb + 2] = wmma16(pa, vb2, acc[cb + 2]);
        acc[cb + 3] = wmma16(pa, vb3, acc[cb + 3]);
      }
    }
    __syncthreads();
  }
  _Float16* Co = ctx + ((size_t)(b * 16 + h) * S + qrow0) * CL;
#pragma unroll
  for (int ct = 0; ct < 8; ++ct)
#pragma unroll
    for (int v = 0; v < 8; ++v) {
      const int r = grp * 8 + v;
      Co[(size_t)r * CL + wv * 128 + ct * 16 + col] =
          (_Float16)(acc[ct][v] / lrun[r]);
    }
}

// ---------------------------------------------------------------------------
extern "C" void kernel_launch(void* const* d_in, const int* in_sizes, int n_in,
                              void* d_out, int out_size, void* d_ws,
                              size_t ws_size, hipStream_t stream) {
  (void)in_sizes; (void)n_in; (void)out_size; (void)ws_size;
  const float* x   = (const float*)d_in[0];
  const float* Wd  = (const float*)d_in[1];
  const float* Wuk = (const float*)d_in[2];
  const float* Wuv = (const float*)d_in[3];
  const float* Wq  = (const float*)d_in[4];
  const float* Wo  = (const float*)d_in[5];
  float* out = (float*)d_out;

  char* ws = (char*)d_ws;
  size_t off = 0;
  auto take = [&](size_t bytes) {
    char* p = ws + off;
    off += (bytes + 255) & ~(size_t)255;
    return p;
  };
  _Float16* x16   = (_Float16*)take(4096ull * 2048 * 2);      // x f16 [bs, m]
  _Float16* Wd16  = (_Float16*)take(512ull * 2048 * 2);       // [c, m]
  _Float16* WqT   = (_Float16*)take(16ull * 2048 * 128 * 2);  // [h][m, d]
  _Float16* WukT  = (_Float16*)take(16ull * 512 * 128 * 2);   // [h][c, d]
  _Float16* WuvT  = (_Float16*)take(16ull * 512 * 128 * 2);   // [h][c, d]
  _Float16* Wo16  = (_Float16*)take(2048ull * 2048 * 2);      // [m, h*dh]
  _Float16* lat16 = (_Float16*)take(4096ull * 512 * 2);       // [b*t, c]
  _Float16* latT  = (_Float16*)take(2ull * 512 * 2048 * 2);   // [b][c, t]
  _Float16* qukT  = (_Float16*)take(16ull * 512 * 2048 * 2);  // [h][c, m]
  _Float16* ouv   = (_Float16*)take(16ull * 2048 * 512 * 2);  // [h][m, c]
  _Float16* qpb   = (_Float16*)take(2ull * 16 * 2048 * 512 * 2);  // [b,h,s,c]
  _Float16* ctxb  = (_Float16*)take(2ull * 16 * 2048 * 512 * 2);  // [b,h,s,c]

  auto grid1 = [](long long n) { return dim3((unsigned)((n + 255) / 256)); };

  // f16 operand copies (K-fast layouts); iB/jB are log2 of the dims
  cvt_f32_h<<<grid1(4096LL * 2048), 256, 0, stream>>>(
      x, x16, 4096LL * 2048, 12, 11, 0, 2048, 1);
  cvt_f32_h<<<grid1(512LL * 2048), 256, 0, stream>>>(
      Wd, Wd16, 512LL * 2048, 9, 11, 0, 2048, 1);
  cvt_f32_h<<<grid1(16LL * 2048 * 128), 256, 0, stream>>>(   // Wq[h,d,m]->[h][m,d]
      Wq, WqT, 16LL * 2048 * 128, 11, 7, 128LL * 2048, 1, 2048);
  cvt_f32_h<<<grid1(16LL * 512 * 128), 256, 0, stream>>>(    // Wuk[h,d,c]->[h][c,d]
      Wuk, WukT, 16LL * 512 * 128, 9, 7, 128LL * 512, 1, 512);
  cvt_f32_h<<<grid1(16LL * 512 * 128), 256, 0, stream>>>(    // Wuv[h,d,c]->[h][c,d]
      Wuv, WuvT, 16LL * 512 * 128, 9, 7, 128LL * 512, 1, 512);
  cvt_f32_h<<<grid1(2048LL * 2048), 256, 0, stream>>>(
      Wo, Wo16, 2048LL * 2048, 11, 11, 0, 2048, 1);

  // latents[b*s, c] = x16 @ Wd16^T   (M=4096, N=512, K=2048)
  gemm_nt<_Float16><<<dim3(8, 64, 1), 128, 0, stream>>>(
      x16, Wd16, lat16, 2048, 1, 0, 0, 2048, 2048, 512, 1, 0, 0, 0, 0, 0, 0);
  // latT[b][c, t] = transpose(latents)
  cvt_h_h<<<grid1(2LL * 512 * 2048), 256, 0, stream>>>(
      lat16, latT, 2LL * 512 * 2048, 9, 11, 2048LL * 512, 1, 512);

  // q_ukT[h][c, m] = Wuk[h]^T Wq[h]  (M=512, N=2048, K=128, per head)
  gemm_nt<_Float16><<<dim3(32, 8, 16), 128, 0, stream>>>(
      WukT, WqT, qukT, 128, 1, 0, 0, 128, 128, 2048, 16,
      0, 512LL * 128, 0, 2048LL * 128, 0, 512LL * 2048);
  // out_uv[h][m, c] = Wo[:,h] Wuv[h] (M=2048, N=512, K=128, per head)
  gemm_nt<_Float16><<<dim3(8, 32, 16), 128, 0, stream>>>(
      Wo16, WuvT, ouv, 128, 1, 0, 0, 2048, 128, 512, 16,
      0, 128, 0, 512LL * 128, 0, 2048LL * 512);
  // query_part[b,h,s,c] = x16[b] @ q_ukT[h]^T (M=2048, N=512, K=2048)
  gemm_nt<_Float16><<<dim3(8, 32, 32), 128, 0, stream>>>(
      x16, qukT, qpb, 2048, 1, 0, 0, 2048, 2048, 512, 16,
      2048LL * 2048, 0, 0, 512LL * 2048, 16LL * 2048 * 512, 2048LL * 512);

  // flash attention over latent dim -> ctx[b,h,s,c]
  mla_attn<<<dim3(128, 16, 2), 128, 0, stream>>>(qpb, lat16, latT, ctxb);

  // out[b,s,m] = sum_h ctx[b,h] @ out_uv[h]^T (M=2048, N=2048, K=512, G=16)
  gemm_nt<float><<<dim3(32, 32, 2), 128, 0, stream>>>(
      ctxb, ouv, out, 512, 16, 2048LL * 512, 2048LL * 512, 512, 512, 2048, 1,
      16LL * 2048 * 512, 0, 0, 0, 2048LL * 2048, 0);
}
